// ConductanceLIFNetwork_IO_20864951124247
// MI455X (gfx1250) — matrix operand order repaired and checked by reading, not compiled, not run on hardware
//
#include <hip/hip_runtime.h>
#include <stdint.h>

typedef __attribute__((ext_vector_type(16))) _Float16 v16h;
typedef __attribute__((ext_vector_type(8)))  _Float16 v8h;
typedef __attribute__((ext_vector_type(8)))  float    v8f;

#define N_NEURONS 1536
#define N_INPUTS  768
#define BATCH     32
#define T_STEPS   256

// workspace layout (bytes); total ~33 MB
#define OFF_WT    0ull                 // Wt[m][n]  f16, 1536x1536
#define OFF_WFT   4718592ull           // WFt[m][k] f16, 1536x768
#define OFF_INPF  7077888ull           // inpF[t][n][b] f16, 256x1536x32
#define OFF_SPK   32243712ull          // 4 buffers of [b][n] f16 (A0,A1,C0,C1)
#define OFF_BAR   32636928ull          // 256 u32 barrier slots
#define SPK_ELEMS (BATCH * N_NEURONS)  // 49152
#define SIM_WGS   48
#define LSTR      1552                 // padded LDS row stride (halves)

// ---------------- prep: W^T with scaling, f32 -> f16 ----------------
__global__ void prep_wt_kernel(const float* __restrict__ w,
                               const float* __restrict__ scal,
                               const int* __restrict__ ct,
                               _Float16* __restrict__ WT) {
  for (size_t i = (size_t)blockIdx.x * blockDim.x + threadIdx.x;
       i < (size_t)N_NEURONS * N_NEURONS; i += (size_t)gridDim.x * blockDim.x) {
    int m = (int)(i / N_NEURONS);
    int nn = (int)(i % N_NEURONS);
    WT[i] = (_Float16)(w[(size_t)nn * N_NEURONS + m] * scal[ct[nn] * 2 + ct[m]]);
  }
}

__global__ void prep_wft_kernel(const float* __restrict__ wff,
                                const float* __restrict__ scalF,
                                const int* __restrict__ ct,
                                _Float16* __restrict__ WFT) {
  for (size_t i = (size_t)blockIdx.x * blockDim.x + threadIdx.x;
       i < (size_t)N_NEURONS * N_INPUTS; i += (size_t)gridDim.x * blockDim.x) {
    int m = (int)(i / N_INPUTS);
    int k = (int)(i % N_INPUTS);
    WFT[i] = (_Float16)(wff[(size_t)k * N_NEURONS + m] * scalF[ct[m]]);
  }
}

__global__ void prep_zero_kernel(_Float16* __restrict__ spk,
                                 unsigned* __restrict__ bar) {
  const size_t total = 4ull * SPK_ELEMS + 256;
  for (size_t i = (size_t)blockIdx.x * blockDim.x + threadIdx.x; i < total;
       i += (size_t)gridDim.x * blockDim.x) {
    if (i < 4ull * SPK_ELEMS) spk[i] = (_Float16)0.0f;
    else                      bar[i - 4ull * SPK_ELEMS] = 0u;
  }
}

// -------- FF GEMM (hoisted out of the sequential loop) --------------
// inpF[t][n][b] = sum_k I[b][t][k] * WF[k][n]*sfF ; [8192x768]@[768x1536]
__global__ void __launch_bounds__(256)
ff_gemm_kernel(const float* __restrict__ inp,      // [32][256][768] f32 spikes
               const _Float16* __restrict__ WFT,   // [1536][768]
               _Float16* __restrict__ inpF) {      // [256][1536][32]
  const int tid  = threadIdx.x;
  const int lane = tid & 31;
  const int gw   = blockIdx.x * 8 + (tid >> 5);    // global wave id, 49152 total
  const int tileN = gw % (N_NEURONS / 16);
  const int tileM = gw / (N_NEURONS / 16);
  const int M0   = tileM * 16;                     // row base in (t*32+b) space
  const int t0   = M0 >> 5;
  const int brel = M0 & 31;
  const int col  = tileN * 16 + (lane & 15);
  const int gm   = M0 + (lane & 15);               // A-fragment row
  const int ab   = gm & 31;                        // batch
  const int at   = gm >> 5;                        // timestep
  const int c0a  = (lane < 16) ? 0 : 16;
  const int c0b  = (lane < 16) ? 0 : 16;

  const float*    arow = inp + ((size_t)ab * T_STEPS + at) * N_INPUTS;
  const _Float16* brow = WFT + (size_t)col * N_INPUTS;

  v8f acc = {};
  for (int kk = 0; kk < N_INPUTS; kk += 32) {
    union { v16h v; _Float16 h[16]; } a;
    const float* p0 = arow + kk + c0a;
#pragma unroll
    for (int i = 0; i < 8; ++i) a.h[i]     = (_Float16)p0[i];
#pragma unroll
    for (int i = 0; i < 8; ++i) a.h[8 + i] = (_Float16)p0[8 + i];
    v16h bf = *(const v16h*)(brow + kk + c0b);
    acc = __builtin_amdgcn_wmma_f32_16x16x32_f16(false, a.v, false, bf,
                                                 (short)0, acc, false, false);
  }
  union { v8h v; _Float16 h[8]; } d;
#pragma unroll
  for (int e = 0; e < 8; ++e) d.h[e] = (_Float16)acc[e];
  *(v8h*)(inpF + ((size_t)t0 * N_NEURONS + col) * BATCH + brel +
          ((lane < 16) ? 0 : 8)) = d.v;
}

// ------------- persistent sim kernel: 48 WGs x 128 threads ----------
__global__ void __launch_bounds__(128)
sim_kernel(const _Float16* __restrict__ WT,    // [1536][1536] (Wt[m][n])
           const _Float16* __restrict__ inpF,  // [256][1536][32]
           _Float16* __restrict__ spk,         // A0,A1,C0,C1 each [32][1536]
           unsigned* __restrict__ bar,         // [256]
           const int* __restrict__ ct,         // [1536]
           float* __restrict__ out) {          // spk[32][256][1536] | volts[...]
  extern __shared__ char smem_raw[];
  _Float16* lds = (_Float16*)smem_raw;

  const int tid   = threadIdx.x;
  const int lane  = tid & 31;
  const int wid   = tid >> 5;
  const int rbase = (wid >> 1) * 16;                 // batch-tile base (0|16)
  const int col_l = (wid & 1) * 16 + (lane & 15);    // local column 0..31
  const int n     = blockIdx.x * 32 + col_l;         // global neuron
  const int roff  = (lane < 16) ? 0 : 8;             // D-fragment row offset
  const int c0a   = (lane < 16) ? 0 : 16;            // A-fragment K offset
  const int c0b   = (lane < 16) ? 0 : 16;            // B-fragment K offset
  const int b_a   = rbase + (lane & 15);             // A-fragment batch row

  // stage this WG's W^T slice (32 cols x 1536 k, padded rows) into LDS once
  for (int idx = tid; idx < 32 * 192; idx += blockDim.x) {
    int cl  = idx / 192;
    int off = (idx % 192) * 8;
    *(v8h*)(lds + cl * LSTR + off) =
        *(const v8h*)(WT + (size_t)(blockIdx.x * 32 + cl) * N_NEURONS + off);
  }
  __syncthreads();

  // per-lane neuron constants (column n fixed per lane)
  const int ctn = ct[n];
  const float leak   = ctn ? 0.01f : 0.005f;  // DT/(tau_mem*g_l)
  const float refstp = ctn ? 1.0f : 2.0f;     // tau_ref/DT
  const float th = -50.f, ur = -65.f, el = -65.f, gl = 10.f;
  const float ar0 = 0.13533528323661270f;  // exp(-1/0.5)
  const float ar1 = 0.60653065971263342f;  // exp(-1/2)
  const float ar2 = 0.13533528323661270f;
  const float ad0 = 0.60653065971263342f;
  const float ad1 = 0.99004983374916811f;  // exp(-1/100)
  const float ad2 = 0.81873075307798186f;  // exp(-1/5)
  const float arF = 0.13533528323661270f;
  const float adF = 0.60653065971263342f;

  float U[8], x0[8], x1[8], x2[8], g0[8], g1[8], g2[8], xF[8], gF[8], rf[8];
#pragma unroll
  for (int e = 0; e < 8; ++e) {
    U[e] = el; x0[e] = x1[e] = x2[e] = 0.f; g0[e] = g1[e] = g2[e] = 0.f;
    xF[e] = gF[e] = rf[e] = 0.f;
  }

  float* outS = out;
  float* outV = out + (size_t)BATCH * T_STEPS * N_NEURONS;
  const size_t arow = (size_t)b_a * N_NEURONS;

  for (int t = 0; t < T_STEPS; ++t) {
    const int p = t & 1;
    const _Float16* sAr = spk + (size_t)p * SPK_ELEMS;
    const _Float16* sCr = spk + (size_t)(2 + p) * SPK_ELEMS;
    _Float16* sAw = spk + (size_t)(1 - p) * SPK_ELEMS;
    _Float16* sCw = spk + (size_t)(3 - p) * SPK_ELEMS;

    v8f GA = {};  // input from type-0 presynaptics (synapses 0 and 1)
    v8f GC = {};  // input from type-1 presynaptics (synapse 2)
#pragma unroll 4
    for (int kk = 0; kk < N_NEURONS; kk += 32) {
      union { v16h v; v8h h[2]; } aA, aC;
      aA.h[0] = *(const v8h*)(sAr + arow + kk + c0a);
      aA.h[1] = *(const v8h*)(sAr + arow + kk + c0a + 8);
      aC.h[0] = *(const v8h*)(sCr + arow + kk + c0a);
      aC.h[1] = *(const v8h*)(sCr + arow + kk + c0a + 8);
      v16h bf = *(const v16h*)(lds + col_l * LSTR + kk + c0b);
      GA = __builtin_amdgcn_wmma_f32_16x16x32_f16(false, aA.v, false, bf,
                                                  (short)0, GA, false, false);
      GC = __builtin_amdgcn_wmma_f32_16x16x32_f16(false, aC.v, false, bf,
                                                  (short)0, GC, false, false);
    }

    union { v8h v; _Float16 h[8]; } fF;
    fF.v = *(const v8h*)(inpF + ((size_t)t * N_NEURONS + n) * BATCH + rbase + roff);

#pragma unroll
    for (int e = 0; e < 8; ++e) {
      const float iA = GA[e], iC = GC[e], iF = (float)fF.h[e];
      x0[e] = ar0 * x0[e] + iA;  g0[e] = ad0 * g0[e] + x0[e];
      x1[e] = ar1 * x1[e] + iA;  g1[e] = ad1 * g1[e] + x1[e];
      x2[e] = ar2 * x2[e] + iC;  g2[e] = ad2 * g2[e] + x2[e];
      xF[e] = arF * xF[e] + iF;  gF[e] = adF * gF[e] + xF[e];
      const float gtot = g0[e] + 0.5f * g1[e] + g2[e] + gF[e];
      const float gE   = -70.f * g2[e];           // gbar*Erev: only synapse 2
      const float Isyn = gE - gtot * U[e];
      float Un = U[e] + leak * (gl * (el - U[e]) + Isyn);
      Un = (rf[e] > 0.f) ? ur : Un;               // refractory clamp
      rf[e] = fmaxf(rf[e] - 1.f, 0.f);
      const float sv = (Un >= th) ? 1.f : 0.f;
      const float Uo = (sv > 0.f) ? ur : Un;
      rf[e] = (sv > 0.f) ? refstp : rf[e];
      U[e] = Uo;

      const int b = rbase + roff + e;
      const size_t oi = ((size_t)b * T_STEPS + t) * N_NEURONS + n;
      __builtin_nontemporal_store(sv, &outS[oi]);
      __builtin_nontemporal_store(Uo, &outV[oi]);
      const _Float16 hs = (_Float16)sv;
      const _Float16 hz = (_Float16)0.f;
      sAw[(size_t)b * N_NEURONS + n] = ctn ? hz : hs;
      sCw[(size_t)b * N_NEURONS + n] = ctn ? hs : hz;
    }

    // grid-wide release/acquire barrier (one single-use slot per step)
    __threadfence();
    __syncthreads();
    if (tid == 0) {
      __hip_atomic_fetch_add(&bar[t], 1u, __ATOMIC_ACQ_REL,
                             __HIP_MEMORY_SCOPE_AGENT);
      while (__hip_atomic_load(&bar[t], __ATOMIC_ACQUIRE,
                               __HIP_MEMORY_SCOPE_AGENT) < (unsigned)SIM_WGS) {
        __builtin_amdgcn_s_sleep(1);
      }
    }
    __syncthreads();
  }
}

extern "C" void kernel_launch(void* const* d_in, const int* in_sizes, int n_in,
                              void* d_out, int out_size, void* d_ws,
                              size_t ws_size, hipStream_t stream) {
  (void)in_sizes; (void)n_in; (void)out_size; (void)ws_size;
  const float* inp   = (const float*)d_in[0];  // [32][256][768]
  const float* w     = (const float*)d_in[1];  // [1536][1536]
  const float* wff   = (const float*)d_in[2];  // [768][1536]
  const float* scal  = (const float*)d_in[3];  // [2][2]
  const float* scalF = (const float*)d_in[4];  // [1][2]
  const int*   ct    = (const int*)d_in[5];    // [1536]
  // d_in[6]: cell_type_indices_FF (all zero) — folded into scalF indexing

  char* ws = (char*)d_ws;
  _Float16* WT   = (_Float16*)(ws + OFF_WT);
  _Float16* WFT  = (_Float16*)(ws + OFF_WFT);
  _Float16* inpF = (_Float16*)(ws + OFF_INPF);
  _Float16* spk  = (_Float16*)(ws + OFF_SPK);
  unsigned* bar  = (unsigned*)(ws + OFF_BAR);
  float* out = (float*)d_out;

  prep_wt_kernel<<<2048, 256, 0, stream>>>(w, scal, ct, WT);
  prep_wft_kernel<<<1024, 256, 0, stream>>>(wff, scalF, ct, WFT);
  prep_zero_kernel<<<256, 256, 0, stream>>>(spk, bar);
  ff_gemm_kernel<<<6144, 256, 0, stream>>>(inp, WFT, inpF);
  sim_kernel<<<SIM_WGS, 128, 32 * LSTR * 2, stream>>>(WT, inpF, spk, bar, ct, out);
}
